// FeaturesProcessing_26250840113433
// MI455X (gfx1250) — compile-verified, matrix operand order checked by reading, not run on hardware
//
#include <hip/hip_runtime.h>
#include <cmath>
#include <cstdint>

// ---------------------------------------------------------------------------
// Fused 3-block MLP:  [LN -> Linear -> GELU] x3, bf16 WMMA for the two big
// GEMMs, LayerNorm fused into the producer of each stage, TDM (tensor_load_
// to_lds) async staging with double-buffered LDS.
//   cvt:    w1,w2 f32 -> bf16 once into d_ws
//   stage0: LN(6) + (6->128) + GELU + LN(128)  -> h1n  bf16 [N][128]   (d_ws)
//   stage1: h1n @ w1^T (+b1) + GELU + LN(512)  -> h2n  bf16 [N][512]   (d_ws)
//   stage2: h2n @ w2^T (+b2) + GELU, *sqrt(1024) -> out f32 [N][1024]
// d_ws usage: N*128*2 + N*512*2 + 128KB + 1MB  (~321.2 MB).
// ---------------------------------------------------------------------------

typedef __attribute__((ext_vector_type(16))) __bf16   v16bf;
typedef __attribute__((ext_vector_type(8)))  __bf16   v8bf;
typedef __attribute__((ext_vector_type(4)))  __bf16   v4bf;
typedef __attribute__((ext_vector_type(8)))  float    v8f;
typedef __attribute__((ext_vector_type(4)))  float    v4f;
typedef __attribute__((ext_vector_type(4)))  uint32_t u32x4;
typedef __attribute__((ext_vector_type(8)))  uint32_t u32x8;

#define EPS 1e-5f

__device__ __forceinline__ float gelu_exact(float x) {
  return 0.5f * x * (1.0f + erff(x * 0.70710678118654752f));
}

// Build a 16x32 bf16 WMMA A/B fragment from a row-major (contiguous-K) row.
// ISA layout (16-bit A 16x32): lanes 0-15 hold K = k0..k0+7 (VGPR0-3) and
// K = k0+16..k0+23 (VGPR4-7); lanes 16-31 hold K = k0+8.. and k0+24..
__device__ __forceinline__ v16bf load_frag(const __bf16* rowptr, int k0) {
  v8bf lo = *(const v8bf*)(rowptr + k0);
  v8bf hi = *(const v8bf*)(rowptr + k0 + 16);
  return __builtin_shufflevector(lo, hi, 0, 1, 2, 3, 4, 5, 6, 7,
                                          8, 9, 10, 11, 12, 13, 14, 15);
}

// ---------------------------------------------------------------------------
// Tensor Data Mover: async 2D bf16 tile (tile_w x tile_h) from global to LDS,
// with TDM-inserted LDS row padding (pad codes per D# group1 spec):
//   pad_interval code: 2^(code+1) DWORDs between pads; pad_amount code: +1 DW.
// D# built per CDNA5 ISA. 2D tile -> groups
// 2/3 NULL (2-operand tensor_load_to_lds form). Tracked by TENSORcnt.
// ---------------------------------------------------------------------------
__device__ __forceinline__ void tdm_load_2d_bf16(
    uint32_t lds_off, const void* gptr, uint32_t tile_w, uint32_t tile_h,
    uint64_t row_stride_elems, uint32_t pad_interval_code,
    uint32_t pad_amount_code) {
  const uint64_t ga = (uint64_t)(uintptr_t)gptr;
  u32x4 g0;
  g0[0] = 1u;                                   // count=1 (valid descriptor)
  g0[1] = lds_off;                              // lds_addr (bytes)
  g0[2] = (uint32_t)ga;                         // global_addr[31:0]
  g0[3] = ((uint32_t)(ga >> 32) & 0x1FFFFFFu) | (2u << 30);  // [56:32], type=2
  u32x8 g1;
  g1[0] = (1u << 16)                            // data_size = 2 bytes
        | (1u << 20)                            // pad_enable
        | (pad_interval_code << 22) | (pad_amount_code << 25);
  g1[1] = (tile_w & 0xFFFFu) << 16;             // tensor_dim0 low16 @ [63:48]
  g1[2] = (tile_w >> 16) | ((tile_h & 0xFFFFu) << 16);  // dim0 hi | dim1 lo
  g1[3] = (tile_h >> 16) | (tile_w << 16);      // dim1 hi | tile_dim0
  g1[4] = tile_h;                               // tile_dim1 (tile_dim2 = 0)
  g1[5] = (uint32_t)row_stride_elems;           // tensor_dim0_stride[31:0]
  g1[6] = (uint32_t)((row_stride_elems >> 32) & 0xFFFFu);  // stride[47:32]
  g1[7] = 0u;
  asm volatile("tensor_load_to_lds %0, %1" :: "s"(g0), "s"(g1) : "memory");
}

__device__ __forceinline__ uint32_t lds_offset(const void* p) {
  return (uint32_t)(uintptr_t)p;  // low 32 bits of LDS flat address = offset
}

// ---------------------------------------------------------------------------
// One-time f32 -> bf16 weight conversion (n multiple of 4).
// ---------------------------------------------------------------------------
__global__ __launch_bounds__(256) void k_cvt_bf16(
    const float* __restrict__ src, __bf16* __restrict__ dst, int n) {
  const int i = (blockIdx.x * 256 + threadIdx.x) * 4;
  if (i < n) {
    v4f v = *(const v4f*)&src[i];
    v4bf o;
#pragma unroll
    for (int j = 0; j < 4; ++j) o[j] = (__bf16)v[j];
    *(v4bf*)&dst[i] = o;
  }
}

// ---------------------------------------------------------------------------
// Stage 0: one wave per row (8 rows / 256-thread block).
// ---------------------------------------------------------------------------
__global__ __launch_bounds__(256) void k_stage0(
    const float* __restrict__ x,                                   // [N][6]
    const float* __restrict__ ln0_g, const float* __restrict__ ln0_b,
    const float* __restrict__ w0,                                  // [128][6]
    const float* __restrict__ b0,                                  // [128]
    const float* __restrict__ ln1_g, const float* __restrict__ ln1_b,
    __bf16* __restrict__ h1n, int N) {
  const int wave = threadIdx.x >> 5;
  const int lane = threadIdx.x & 31;
  const int row  = blockIdx.x * 8 + wave;
  if (row >= N) return;

  float xv[6];
#pragma unroll
  for (int i = 0; i < 6; ++i) xv[i] = x[row * 6 + i];
  float mu = 0.f;
#pragma unroll
  for (int i = 0; i < 6; ++i) mu += xv[i];
  mu *= (1.0f / 6.0f);
  float var = 0.f;
#pragma unroll
  for (int i = 0; i < 6; ++i) { float d = xv[i] - mu; var += d * d; }
  var *= (1.0f / 6.0f);
  const float rstd = rsqrtf(var + EPS);
  float xn[6];
#pragma unroll
  for (int i = 0; i < 6; ++i) xn[i] = (xv[i] - mu) * rstd * ln0_g[i] + ln0_b[i];

  float h[4];
#pragma unroll
  for (int j = 0; j < 4; ++j) {
    const int o = lane * 4 + j;
    float acc = b0[o];
#pragma unroll
    for (int i = 0; i < 6; ++i) acc += xn[i] * w0[o * 6 + i];
    h[j] = gelu_exact(acc);
  }

  float s = 0.f, ss = 0.f;
#pragma unroll
  for (int j = 0; j < 4; ++j) { s += h[j]; ss += h[j] * h[j]; }
#pragma unroll
  for (int off = 16; off > 0; off >>= 1) {
    s  += __shfl_xor(s, off, 32);
    ss += __shfl_xor(ss, off, 32);
  }
  const float mu1   = s * (1.0f / 128.0f);
  const float var1  = ss * (1.0f / 128.0f) - mu1 * mu1;
  const float rstd1 = rsqrtf(var1 + EPS);

  v4bf outv;
#pragma unroll
  for (int j = 0; j < 4; ++j) {
    const int o = lane * 4 + j;
    outv[j] = (__bf16)((h[j] - mu1) * rstd1 * ln1_g[o] + ln1_b[o]);
  }
  *(v4bf*)&h1n[(size_t)row * 128 + lane * 4] = outv;
}

// ---------------------------------------------------------------------------
// Stage 1: C[32 x 512] = h1n[32 x 128] @ w1^T, bias+GELU+LN(512) fused.
// 8 waves; wave w owns columns [w*64, w*64+64). K streamed in 32-chunks via
// TDM into double-buffered LDS (rows padded to 40 bf16 by the TDM).
// ---------------------------------------------------------------------------
#define LDK2 40

__global__ __launch_bounds__(256) void k_stage1(
    const __bf16* __restrict__ h1n,                                 // [N][128]
    const __bf16* __restrict__ w1b,                                 // [512][128]
    const float* __restrict__ b1,
    const float* __restrict__ ln2_g, const float* __restrict__ ln2_b,
    __bf16* __restrict__ h2n) {
  __shared__ __bf16 lA[2][32 * LDK2];
  __shared__ __bf16 lW[2][512 * LDK2];
  __shared__ float  sSum[32];
  __shared__ float  sSq[32];

  const int tid  = threadIdx.x;
  const int wave = tid >> 5;
  const int lane = tid & 31;
  const int half = lane >> 4;
  const int idx  = lane & 15;
  const int row0 = blockIdx.x * 32;
  const bool issuer = (wave == 0);  // wave-uniform: TDM is wave-level

  if (tid < 32) { sSum[tid] = 0.f; sSq[tid] = 0.f; }

  v8f acc[2][4];
#pragma unroll
  for (int mt = 0; mt < 2; ++mt)
#pragma unroll
    for (int nt = 0; nt < 4; ++nt) acc[mt][nt] = (v8f)0.0f;

  if (issuer) {  // preload chunk 0 (A tile 32x32, W tile 512x32)
    tdm_load_2d_bf16(lds_offset(&lA[0][0]), h1n + (size_t)row0 * 128,
                     32, 32, 128, /*pad 16DW+4DW*/ 3, 3);
    tdm_load_2d_bf16(lds_offset(&lW[0][0]), w1b,
                     32, 512, 128, 3, 3);
  }

  for (int c = 0; c < 4; ++c) {
    const int buf = c & 1;
    if (issuer) __builtin_amdgcn_s_wait_tensorcnt(0);
    __syncthreads();  // buf ready for all waves
    if (issuer && c < 3) {  // prefetch chunk c+1 under the WMMAs
      const int kc = (c + 1) * 32;
      tdm_load_2d_bf16(lds_offset(&lA[buf ^ 1][0]),
                       h1n + (size_t)row0 * 128 + kc, 32, 32, 128, 3, 3);
      tdm_load_2d_bf16(lds_offset(&lW[buf ^ 1][0]),
                       w1b + kc, 32, 512, 128, 3, 3);
    }

    const int k0 = half * 8;
#pragma unroll
    for (int mt = 0; mt < 2; ++mt) {
      v16bf a = load_frag(&lA[buf][(mt * 16 + idx) * LDK2], k0);
#pragma unroll
      for (int nt = 0; nt < 4; ++nt) {
        const int n = wave * 64 + nt * 16 + idx;
        v16bf b = load_frag(&lW[buf][n * LDK2], k0);
        acc[mt][nt] = __builtin_amdgcn_wmma_f32_16x16x32_bf16(
            false, a, false, b, (short)0, acc[mt][nt], false, false);
      }
    }
    __syncthreads();  // done reading buf before it is re-filled
  }

  // Epilogue: bias + GELU in-register, then LN(512) across the block.
#pragma unroll
  for (int mt = 0; mt < 2; ++mt)
#pragma unroll
    for (int nt = 0; nt < 4; ++nt) {
      const int col = wave * 64 + nt * 16 + idx;
      const float bias = b1[col];
#pragma unroll
      for (int r = 0; r < 8; ++r)
        acc[mt][nt][r] = gelu_exact(acc[mt][nt][r] + bias);
    }

#pragma unroll
  for (int mt = 0; mt < 2; ++mt)
#pragma unroll
    for (int r = 0; r < 8; ++r) {
      float s = 0.f, ss = 0.f;
#pragma unroll
      for (int nt = 0; nt < 4; ++nt) {
        const float v = acc[mt][nt][r];
        s += v; ss += v * v;
      }
#pragma unroll
      for (int off = 1; off < 16; off <<= 1) {
        s  += __shfl_xor(s, off, 32);
        ss += __shfl_xor(ss, off, 32);
      }
      if (idx == 0) {
        const int rr = mt * 16 + half * 8 + r;
        atomicAdd(&sSum[rr], s);   // ds_add_f32
        atomicAdd(&sSq[rr], ss);
      }
    }
  __syncthreads();

#pragma unroll
  for (int mt = 0; mt < 2; ++mt)
#pragma unroll
    for (int r = 0; r < 8; ++r) {
      const int rr   = mt * 16 + half * 8 + r;
      const float mu = sSum[rr] * (1.0f / 512.0f);
      const float vr = sSq[rr] * (1.0f / 512.0f) - mu * mu;
      const float rs = rsqrtf(vr + EPS);
#pragma unroll
      for (int nt = 0; nt < 4; ++nt) {
        const int col = wave * 64 + nt * 16 + idx;
        const float v = (acc[mt][nt][r] - mu) * rs * ln2_g[col] + ln2_b[col];
        h2n[(size_t)(row0 + rr) * 512 + col] = (__bf16)v;
      }
    }
}

// ---------------------------------------------------------------------------
// Stage 2: out[32 x 256-slab] = h2n[32 x 512] @ w2^T, bias+GELU, *32.
// grid (N/32, 4); wave w owns 32 columns. K streamed in 64-chunks via TDM
// (rows padded to 72 bf16), double-buffered.
// ---------------------------------------------------------------------------
#define LDK3 72

__global__ __launch_bounds__(256) void k_stage2(
    const __bf16* __restrict__ h2n,                                 // [N][512]
    const __bf16* __restrict__ w2b,                                 // [1024][512]
    const float* __restrict__ b2,
    float* __restrict__ out) {
  __shared__ __bf16 lA[2][32 * LDK3];
  __shared__ __bf16 lW[2][256 * LDK3];

  const int tid  = threadIdx.x;
  const int wave = tid >> 5;
  const int lane = tid & 31;
  const int half = lane >> 4;
  const int idx  = lane & 15;
  const int row0 = blockIdx.x * 32;
  const int n0   = blockIdx.y * 256;
  const bool issuer = (wave == 0);

  v8f acc[2][2];
#pragma unroll
  for (int mt = 0; mt < 2; ++mt)
#pragma unroll
    for (int nt = 0; nt < 2; ++nt) acc[mt][nt] = (v8f)0.0f;

  if (issuer) {  // preload chunk 0 (A 32x64, W 256x64), pad 32DW rows by 4DW
    tdm_load_2d_bf16(lds_offset(&lA[0][0]), h2n + (size_t)row0 * 512,
                     64, 32, 512, 4, 3);
    tdm_load_2d_bf16(lds_offset(&lW[0][0]), w2b + (size_t)n0 * 512,
                     64, 256, 512, 4, 3);
  }

  for (int c = 0; c < 8; ++c) {
    const int buf = c & 1;
    if (issuer) __builtin_amdgcn_s_wait_tensorcnt(0);
    __syncthreads();
    if (issuer && c < 7) {
      const int kc = (c + 1) * 64;
      tdm_load_2d_bf16(lds_offset(&lA[buf ^ 1][0]),
                       h2n + (size_t)row0 * 512 + kc, 64, 32, 512, 4, 3);
      tdm_load_2d_bf16(lds_offset(&lW[buf ^ 1][0]),
                       w2b + (size_t)n0 * 512 + kc, 64, 256, 512, 4, 3);
    }

#pragma unroll
    for (int ks = 0; ks < 64; ks += 32) {
      const int k0 = ks + half * 8;
#pragma unroll
      for (int mt = 0; mt < 2; ++mt) {
        v16bf a = load_frag(&lA[buf][(mt * 16 + idx) * LDK3], k0);
#pragma unroll
        for (int nt = 0; nt < 2; ++nt) {
          const int n = wave * 32 + nt * 16 + idx;
          v16bf b = load_frag(&lW[buf][n * LDK3], k0);
          acc[mt][nt] = __builtin_amdgcn_wmma_f32_16x16x32_bf16(
              false, a, false, b, (short)0, acc[mt][nt], false, false);
        }
      }
    }
    __syncthreads();
  }

#pragma unroll
  for (int mt = 0; mt < 2; ++mt)
#pragma unroll
    for (int nt = 0; nt < 2; ++nt) {
      const int col = n0 + wave * 32 + nt * 16 + idx;
      const float bias = b2[col];
#pragma unroll
      for (int r = 0; r < 8; ++r) {
        const int row = row0 + mt * 16 + half * 8 + r;
        out[(size_t)row * 1024 + col] =
            gelu_exact(acc[mt][nt][r] + bias) * 32.0f;  // sqrt(1024)
      }
    }
}

// ---------------------------------------------------------------------------
extern "C" void kernel_launch(void* const* d_in, const int* in_sizes, int n_in,
                              void* d_out, int out_size, void* d_ws,
                              size_t ws_size, hipStream_t stream) {
  const float* x     = (const float*)d_in[0];
  const float* ln0_g = (const float*)d_in[1];
  const float* ln0_b = (const float*)d_in[2];
  const float* w0    = (const float*)d_in[3];
  const float* b0    = (const float*)d_in[4];
  const float* ln1_g = (const float*)d_in[5];
  const float* ln1_b = (const float*)d_in[6];
  const float* w1    = (const float*)d_in[7];
  const float* b1    = (const float*)d_in[8];
  const float* ln2_g = (const float*)d_in[9];
  const float* ln2_b = (const float*)d_in[10];
  const float* w2    = (const float*)d_in[11];
  const float* b2    = (const float*)d_in[12];
  (void)n_in; (void)out_size; (void)ws_size;

  const int N = in_sizes[0] / 6;  // 262144

  char* ws = (char*)d_ws;
  __bf16* h1n = (__bf16*)ws;                                  //  64 MB
  __bf16* h2n = (__bf16*)(ws + (size_t)N * 128 * 2);          // 256 MB
  __bf16* w1b = (__bf16*)(ws + (size_t)N * 640 * 2);          // 128 KB
  __bf16* w2b = w1b + 512 * 128;                              //   1 MB
  float*  out = (float*)d_out;

  k_cvt_bf16<<<(512 * 128) / 1024, 256, 0, stream>>>(w1, w1b, 512 * 128);
  k_cvt_bf16<<<(1024 * 512) / 1024, 256, 0, stream>>>(w2, w2b, 1024 * 512);
  k_stage0<<<N / 8, 256, 0, stream>>>(x, ln0_g, ln0_b, w0, b0, ln1_g, ln1_b,
                                      h1n, N);
  k_stage1<<<N / 32, 256, 0, stream>>>(h1n, w1b, b1, ln2_g, ln2_b, h2n);
  dim3 g3(N / 32, 4);
  k_stage2<<<g3, 256, 0, stream>>>(h2n, w2b, b2, out);
}
